// SemanticLayer_48232482734311
// MI455X (gfx1250) — matrix operand order repaired and checked by reading, not compiled, not run on hardware
//
#include <hip/hip_runtime.h>
#include <hip/hip_bf16.h>

typedef __attribute__((ext_vector_type(16))) _Float16 v16h;
typedef __attribute__((ext_vector_type(8)))  _Float16 v8h;
typedef __attribute__((ext_vector_type(8)))  float    v8f;

#define DD    128          // feature dim (K)
#define LL    1024         // sequence length
#define BM    128          // tile rows (x / sentence1)
#define BN    64           // tile cols (y / sentence2)
#define LDST  136          // LDS row stride in halves: 272B = 17*16B -> aligned + bank-rotated

// Fast acos: Abramowitz-Stegun 4.4.45, |err| <= 6.7e-5 over [-1,1].
// acos(x) = sqrt(1-|x|) * poly(|x|), reflected for x < 0.
__device__ __forceinline__ float fast_acos(float v) {
    const float ax = fabsf(v);
    const float r  = sqrtf(1.0f - ax);                    // ax <= 1 guaranteed by clamp
    float p = fmaf(-0.0187292986f, ax, 0.0742610246f);
    p = fmaf(p, ax, -0.2121144314f);
    p = fmaf(p, ax, 1.5707962627f);
    const float ac = r * p;
    return (v < 0.0f) ? (3.14159265358979f - ac) : ac;
}

__global__ __launch_bounds__(256) void semantic_arccos_kernel(
    const float* __restrict__ X,   // [B, LL, DD]
    const float* __restrict__ Y,   // [B, LL, DD]
    float* __restrict__ out)       // [B, LL*LL]
{
    __shared__ _Float16 ldsA[BM * LDST];   // raw f16 rows of X tile
    __shared__ _Float16 ldsB[BN * LDST];   // raw f16 rows of Y tile
    __shared__ float     sA[BM];           // 1/||x_row||
    __shared__ float     sB[BN];           // 1/||y_row||

    const int b  = blockIdx.z;
    const int m0 = blockIdx.y * BM;
    const int n0 = blockIdx.x * BN;
    const int t  = threadIdx.x;

    // ---------------- Stage 1: load rows, sum-of-squares, raw f16 -> LDS ----
    if (t < BM + BN) {
        const bool isB = (t >= BM);
        const int  r   = isB ? (t - BM) : t;
        const float* src = isB ? (Y + ((size_t)b * LL + (n0 + r)) * DD)
                               : (X + ((size_t)b * LL + (m0 + r)) * DD);
        _Float16* dst = (isB ? ldsB : ldsA) + r * LDST;

        float s = 0.f;
        #pragma unroll
        for (int i = 0; i < DD; i += 8) {
            float4 a = *(const float4*)(src + i);
            float4 c = *(const float4*)(src + i + 4);
            s += a.x*a.x + a.y*a.y + a.z*a.z + a.w*a.w;
            s += c.x*c.x + c.y*c.y + c.z*c.z + c.w*c.w;
            v8h h;
            h[0] = (_Float16)a.x; h[1] = (_Float16)a.y;
            h[2] = (_Float16)a.z; h[3] = (_Float16)a.w;
            h[4] = (_Float16)c.x; h[5] = (_Float16)c.y;
            h[6] = (_Float16)c.z; h[7] = (_Float16)c.w;
            *(v8h*)(dst + i) = h;   // 16B ds_store_b128
        }
        const float scale = 1.0f / fmaxf(sqrtf(s), 1e-12f);
        (isB ? sB : sA)[r] = scale;
    }
    __syncthreads();

    // ---------------- Stage 2: WMMA GEMM over K=128 --------------------------
    const int wid  = t >> 5;
    const int lane = t & 31;
    const int half = lane >> 4;      // 0: lanes 0-15, 1: lanes 16-31
    const int mn   = lane & 15;

    const int wm = (wid & 3) * 32;   // wave row offset in tile (4 waves in M)
    const int wn = (wid >> 2) * 32;  // wave col offset in tile (2 waves in N)

    v8f acc[2][2];
    #pragma unroll
    for (int i = 0; i < 2; ++i)
        #pragma unroll
        for (int j = 0; j < 2; ++j)
            acc[i][j] = {};

    #pragma unroll
    for (int k0 = 0; k0 < DD; k0 += 32) {
        v16h afrag[2], bfrag[2];
        // A fragment (16x32 f16): lanes 0-15 hold K 0..7 / 16..23, lanes 16-31 K 8..15 / 24..31
        #pragma unroll
        for (int i = 0; i < 2; ++i) {
            const _Float16* p = ldsA + (wm + i*16 + mn) * LDST + k0 + half*8;
            union { v16h v; v8h p8[2]; } u;
            u.p8[0] = *(const v8h*)(p);        // VGPR0-3
            u.p8[1] = *(const v8h*)(p + 16);   // VGPR4-7
            afrag[i] = u.v;
        }
        // B fragment (32x16 f16): lane = N column, lanes 0-15 K 0..7 / 16..23, lanes 16-31 K 8..15 / 24..31
        #pragma unroll
        for (int j = 0; j < 2; ++j) {
            const _Float16* p = ldsB + (wn + j*16 + mn) * LDST + k0 + half*8;
            union { v16h v; v8h p8[2]; } u;
            u.p8[0] = *(const v8h*)(p);        // K +0..7
            u.p8[1] = *(const v8h*)(p + 16);   // K +16..23
            bfrag[j] = u.v;
        }
        #pragma unroll
        for (int i = 0; i < 2; ++i)
            #pragma unroll
            for (int j = 0; j < 2; ++j)
                acc[i][j] = __builtin_amdgcn_wmma_f32_16x16x32_f16(
                    /*neg_a=*/false, afrag[i],
                    /*neg_b=*/false, bfrag[j],
                    /*c_mod=*/(short)0, acc[i][j],
                    /*reuse_a=*/false, /*reuse_b=*/false);
    }

    // ---------------- Epilogue: scale, clamp/nudge, 1 - acos(x)/pi -----------
    const float INV_PI = 0.31830988618379067154f;
    const size_t outBase = (size_t)b * LL * LL;

    #pragma unroll
    for (int i = 0; i < 2; ++i) {
        #pragma unroll
        for (int j = 0; j < 2; ++j) {
            const int colLocal = wn + j*16 + mn;
            const float sy = sB[colLocal];
            const int col  = n0 + colLocal;
            #pragma unroll
            for (int r = 0; r < 8; ++r) {
                const int rowLocal = wm + i*16 + half*8 + r;  // C/D layout: VGPR r -> M=r (+8 for hi half)
                const float sx = sA[rowLocal];
                float v = acc[i][j][r] * sx * sy;
                v = fminf(fmaxf(v, -1.0f), 1.0f);
                if (v >= 1.0f)       v -= 1e-6f;              // reference nudges, not clamps
                else if (v <= -1.0f) v += 1e-6f;
                const float o = 1.0f - fast_acos(v) * INV_PI;
                out[outBase + (size_t)(m0 + rowLocal) * LL + col] = o;
            }
        }
    }
}

extern "C" void kernel_launch(void* const* d_in, const int* in_sizes, int n_in,
                              void* d_out, int out_size, void* d_ws, size_t ws_size,
                              hipStream_t stream) {
    const float* X = (const float*)d_in[0];
    const float* Y = (const float*)d_in[1];
    float* out = (float*)d_out;

    const int B = in_sizes[0] / (LL * DD);   // 32 for the reference shapes

    dim3 grid(LL / BN, LL / BM, B);          // (16, 8, 32)
    semantic_arccos_kernel<<<grid, 256, 0, stream>>>(X, Y, out);
}
